// AdjointManifold_36859409335063
// MI455X (gfx1250) — matrix operand order repaired and checked by reading, not compiled, not run on hardware
//
#include <hip/hip_runtime.h>

#define DIM   256
#define RANK  32
#define DEPTH 4
#define SEQ   512
#define BATCH 8
#define VOCAB 32000
#define NTILES (VOCAB / 16)   // 2000
#define KSTEPS (DIM / 32)     // 8

typedef __attribute__((ext_vector_type(16))) __bf16 v16bf;
typedef __attribute__((ext_vector_type(8)))  float  v8f;

// ---------------- bf16 split helpers ----------------
__device__ __forceinline__ unsigned short f32_to_bf16(float f) {
  unsigned u = __float_as_uint(f);
  unsigned r = u + 0x7FFFu + ((u >> 16) & 1u);   // round-to-nearest-even
  return (unsigned short)(r >> 16);
}
__device__ __forceinline__ float bf16_to_f32(unsigned short h) {
  return __uint_as_float(((unsigned)h) << 16);
}
__device__ __forceinline__ v16bf castFrag(uint4 a, uint4 b) {
  union { uint4 q[2]; v16bf v; } u;
  u.q[0] = a; u.q[1] = b;
  return u.v;
}

// =====================================================================
// Kernel 1: pack Wr [256,32000] f32 into WMMA B-fragment layout, split
// into bf16 hi/lo planes.  B (K=32 x N=16), this file's contiguous k order (not AMD's table order): lanes 0-15 hold
// N=lane, K=0..15 (VGPR v = K{2v,2v+1}); lanes 16-31 hold K=16..31.
// Linear layout: pk[((ntile*8 + kstep)*32 + lane)*8 + v]
// =====================================================================
__global__ __launch_bounds__(256) void pack_wr_kernel(
    const float* __restrict__ Wr,
    unsigned* __restrict__ pkHi, unsigned* __restrict__ pkLo) {
  int idx  = blockIdx.x * 256 + threadIdx.x;   // < NTILES*8*32*8 = 4,096,000
  int v    = idx & 7;
  int lane = (idx >> 3) & 31;
  int k    = (idx >> 8) & 7;
  int nt   = idx >> 11;
  int n  = nt * 16 + (lane & 15);
  int ka = k * 32 + ((lane < 16) ? 0 : 16) + 2 * v;
  float f0 = Wr[(size_t)ka * VOCAB + n];
  float f1 = Wr[(size_t)(ka + 1) * VOCAB + n];
  unsigned short h0 = f32_to_bf16(f0), h1 = f32_to_bf16(f1);
  float l0 = f0 - bf16_to_f32(h0), l1 = f1 - bf16_to_f32(h1);
  pkHi[idx] = ((unsigned)h1 << 16) | (unsigned)h0;
  pkLo[idx] = ((unsigned)f32_to_bf16(l1) << 16) | (unsigned)f32_to_bf16(l0);
}

// =====================================================================
// Kernel 2: sequential ODE scan. One block (256 thr, 8 waves) per batch
// chain. Thread d owns x[d], v[d]. Stage-1 weights (64/thread) and Wc
// column (32/thread) live in registers, reloaded per layer (L2-hot).
// LDS only for reductions.
// =====================================================================
__device__ __forceinline__ float blockSum(float val, float* red, int tid) {
  #pragma unroll
  for (int off = 16; off > 0; off >>= 1) val += __shfl_down(val, off, 32);
  __syncthreads();                     // protect red[] from previous readers
  if ((tid & 31) == 0) red[tid >> 5] = val;
  __syncthreads();
  float s = red[0];
  #pragma unroll
  for (int w = 1; w < 8; ++w) s += red[w];
  return s;
}

__device__ __forceinline__ float gammaEval(
    float vv, const float (&wreg)[64], const float (&wcr)[32],
    float* vbuf, float* pbuf, float* ab, float* pr, int tid) {
  vbuf[tid] = vv;
  __syncthreads();
  // stage 1: thread computes partial of output o=tid&63 over d-range q*64..
  float part = 0.f;
  const float4* vb4 = (const float4*)(vbuf + ((tid >> 6) << 6));
  #pragma unroll
  for (int j4 = 0; j4 < 16; ++j4) {
    float4 v4 = vb4[j4];
    part = fmaf(v4.x, wreg[4 * j4 + 0], part);
    part = fmaf(v4.y, wreg[4 * j4 + 1], part);
    part = fmaf(v4.z, wreg[4 * j4 + 2], part);
    part = fmaf(v4.w, wreg[4 * j4 + 3], part);
  }
  pbuf[tid] = part;
  __syncthreads();
  if (tid < 64) ab[tid] = pbuf[tid] + pbuf[tid + 64] + pbuf[tid + 128] + pbuf[tid + 192];
  __syncthreads();
  if (tid < 32) pr[tid] = ab[tid] * ab[tid + 32];   // (v@U)*(v@V)
  __syncthreads();
  // stage 2: gamma[d=tid] = sum_r pr[r] * Wc[r][d]
  float g = 0.f;
  const float4* pr4 = (const float4*)pr;
  #pragma unroll
  for (int j4 = 0; j4 < 8; ++j4) {
    float4 p = pr4[j4];
    g = fmaf(p.x, wcr[4 * j4 + 0], g);
    g = fmaf(p.y, wcr[4 * j4 + 1], g);
    g = fmaf(p.z, wcr[4 * j4 + 2], g);
    g = fmaf(p.w, wcr[4 * j4 + 3], g);
  }
  return g;
}

__global__ __launch_bounds__(256) void scan_kernel(
    const int* __restrict__ ids, const float* __restrict__ emb,
    const float* __restrict__ x0v, const float* __restrict__ v0v,
    const float* __restrict__ lnG, const float* __restrict__ lnB,
    const float* __restrict__ U, const float* __restrict__ V,
    const float* __restrict__ Wc,
    const float* __restrict__ roG, const float* __restrict__ roB,
    float* __restrict__ outLN, float* __restrict__ xvOut) {
  __shared__ __align__(16) float vbuf[DIM];
  __shared__ __align__(16) float pbuf[DIM];
  __shared__ __align__(16) float ab[2 * RANK];
  __shared__ __align__(16) float pr[RANK];
  __shared__ __align__(16) float red[8];

  const int tid = threadIdx.x;
  const int b   = blockIdx.x;
  const int o   = tid & 63;     // which of the 64 stage-1 outputs
  const int q   = tid >> 6;     // which K-quarter
  const int r   = o & 31;

  float xd = x0v[tid];
  float vd = v0v[tid];
  float lng[DEPTH], lnb[DEPTH];
  #pragma unroll
  for (int i = 0; i < DEPTH; ++i) {
    lng[i] = lnG[i * DIM + tid];
    lnb[i] = lnB[i * DIM + tid];
  }
  const float rg = roG[tid], rb = roB[tid];
  const float h = 0.25f, h2 = 0.125f, h6 = 0.25f / 6.0f;

  for (int t = 0; t < SEQ; ++t) {
    int tok = ids[b * SEQ + t];
    vd += emb[(size_t)tok * DIM + tid];
    if (t + 1 < SEQ) {
      int tokN = ids[b * SEQ + t + 1];
      __builtin_prefetch(&emb[(size_t)tokN * DIM + tid], 0, 0);
    }
    for (int i = 0; i < DEPTH; ++i) {
      // reload this layer's weight slices into registers (L2-resident)
      const float* Wsel = ((o < RANK) ? U : V) + (size_t)i * (DIM * RANK);
      float wreg[64];
      #pragma unroll
      for (int j = 0; j < 64; ++j) wreg[j] = Wsel[(q * 64 + j) * RANK + r];
      float wcr[RANK];
      #pragma unroll
      for (int j = 0; j < RANK; ++j) wcr[j] = Wc[((size_t)i * RANK + j) * DIM + tid];

      // layernorm(x)
      float m  = blockSum(xd, red, tid) * (1.0f / DIM);
      float cc = xd - m;
      float var = blockSum(cc * cc, red, tid) * (1.0f / DIM);
      xd = fmaf(cc * rsqrtf(var + 1e-5f), lng[i], lnb[i]);

      // RK4: dx/dt = v, dv/dt = -gamma(v)
      for (int s = 0; s < 4; ++s) {
        float g1 = gammaEval(vd, wreg, wcr, vbuf, pbuf, ab, pr, tid);
        float v2 = fmaf(-h2, g1, vd);
        float g2 = gammaEval(v2, wreg, wcr, vbuf, pbuf, ab, pr, tid);
        float v3 = fmaf(-h2, g2, vd);
        float g3 = gammaEval(v3, wreg, wcr, vbuf, pbuf, ab, pr, tid);
        float v4 = fmaf(-h, g3, vd);
        float g4 = gammaEval(v4, wreg, wcr, vbuf, pbuf, ab, pr, tid);
        xd = fmaf(h6, vd + 2.0f * (v2 + v3) + v4, xd);
        vd = fmaf(-h6, g1 + 2.0f * (g2 + g3) + g4, vd);
      }
    }
    // readout layernorm (does not modify x)
    float m  = blockSum(xd, red, tid) * (1.0f / DIM);
    float cc = xd - m;
    float var = blockSum(cc * cc, red, tid) * (1.0f / DIM);
    float on = fmaf(cc * rsqrtf(var + 1e-5f), rg, rb);
    outLN[((size_t)b * SEQ + t) * DIM + tid] = on;
  }
  xvOut[b * DIM + tid] = xd;                 // final x
  xvOut[BATCH * DIM + b * DIM + tid] = vd;   // final v
}

// =====================================================================
// Kernel 3: logits = outLN[4096,256] @ Wr[256,32000] + br, split-bf16
// WMMA with f32 accumulation.  Block = 256 thr / 8 waves, M-chunk = 128
// rows (8 m-tiles, A fragments staged in LDS), wave w owns n-tile
// blockIdx.y*8 + w.  Grid (32, 250) covers everything exactly.
// =====================================================================
__global__ __launch_bounds__(256) void gemm_kernel(
    const float* __restrict__ A, const unsigned* __restrict__ pkHi,
    const unsigned* __restrict__ pkLo, const float* __restrict__ br,
    float* __restrict__ out) {
  extern __shared__ unsigned gsm[];
  unsigned* aHi = gsm;           // 8mt*8k*32lane*8v = 16384 dwords = 64KB
  unsigned* aLo = gsm + 16384;   // 64KB

  const int tid = threadIdx.x;
  const int m0  = blockIdx.x * 128;

  // phase 1: build A fragments (hi/lo bf16) in LDS.
  // A-fragment, this file's contiguous k order (same as B): lane<16 -> M=lane,  K = 0..15
  //                             lane>=16 -> M=lane-16, K = 16..31
  for (int idx = tid; idx < 16384; idx += 256) {
    int v    = idx & 7;
    int lane = (idx >> 3) & 31;
    int k    = (idx >> 8) & 7;
    int mt   = idx >> 11;
    int row = m0 + mt * 16 + (lane & 15);
    int ka  = k * 32 + ((lane < 16) ? 0 : 16) + 2 * v;
    float2 f = *(const float2*)(A + (size_t)row * DIM + ka);
    unsigned short h0 = f32_to_bf16(f.x), h1 = f32_to_bf16(f.y);
    float l0 = f.x - bf16_to_f32(h0), l1 = f.y - bf16_to_f32(h1);
    aHi[idx] = ((unsigned)h1 << 16) | (unsigned)h0;
    aLo[idx] = ((unsigned)f32_to_bf16(l1) << 16) | (unsigned)f32_to_bf16(l0);
  }
  __syncthreads();

  const int w     = tid >> 5;
  const int lane  = tid & 31;
  const int ntile = blockIdx.y * 8 + w;     // < 2000 always

  v8f acc[8];
  #pragma unroll
  for (int mt = 0; mt < 8; ++mt)
    #pragma unroll
    for (int j = 0; j < 8; ++j) acc[mt][j] = 0.0f;

  for (int k = 0; k < KSTEPS; ++k) {
    const uint4* bh = (const uint4*)(pkHi + (((size_t)ntile * 8 + k) * 32 + lane) * 8);
    const uint4* bl = (const uint4*)(pkLo + (((size_t)ntile * 8 + k) * 32 + lane) * 8);
    v16bf bhi = castFrag(bh[0], bh[1]);
    v16bf blo = castFrag(bl[0], bl[1]);
    #pragma unroll
    for (int mt = 0; mt < 8; ++mt) {
      const uint4* ah = (const uint4*)(aHi + ((mt * 8 + k) * 32 + lane) * 8);
      const uint4* al = (const uint4*)(aLo + ((mt * 8 + k) * 32 + lane) * 8);
      v16bf ahi = castFrag(ah[0], ah[1]);
      v16bf alo = castFrag(al[0], al[1]);
      acc[mt] = __builtin_amdgcn_wmma_f32_16x16x32_bf16(false, ahi, false, bhi,
                                                        (short)0, acc[mt], false, false);
      acc[mt] = __builtin_amdgcn_wmma_f32_16x16x32_bf16(false, ahi, false, blo,
                                                        (short)0, acc[mt], false, false);
      acc[mt] = __builtin_amdgcn_wmma_f32_16x16x32_bf16(false, alo, false, bhi,
                                                        (short)0, acc[mt], false, false);
    }
  }

  // store: C/D layout -> VGPR j: lanes 0-15 row m+j, lanes 16-31 row m+8+j
  const int col = ntile * 16 + (lane & 15);
  const float bias = br[col];
  #pragma unroll
  for (int mt = 0; mt < 8; ++mt) {
    int rowBase = m0 + mt * 16 + ((lane < 16) ? 0 : 8);
    #pragma unroll
    for (int j = 0; j < 8; ++j) {
      out[(size_t)(rowBase + j) * VOCAB + col] = acc[mt][j] + bias;
    }
  }
}

// =====================================================================
extern "C" void kernel_launch(void* const* d_in, const int* in_sizes, int n_in,
                              void* d_out, int out_size, void* d_ws, size_t ws_size,
                              hipStream_t stream) {
  (void)in_sizes; (void)n_in; (void)out_size; (void)ws_size;
  const int*   ids = (const int*)d_in[0];
  const float* emb = (const float*)d_in[1];
  const float* x0  = (const float*)d_in[2];
  const float* v0  = (const float*)d_in[3];
  const float* lnG = (const float*)d_in[4];
  const float* lnB = (const float*)d_in[5];
  const float* U   = (const float*)d_in[6];
  const float* V   = (const float*)d_in[7];
  const float* Wc  = (const float*)d_in[8];
  const float* roG = (const float*)d_in[9];
  const float* roB = (const float*)d_in[10];
  const float* Wr  = (const float*)d_in[11];
  const float* br  = (const float*)d_in[12];

  float* out = (float*)d_out;
  float* xv  = out + (size_t)BATCH * SEQ * VOCAB;   // final x then v

  float*    ws    = (float*)d_ws;
  float*    outLN = ws;                                        // 4096*256 f32
  unsigned* pkHi  = (unsigned*)(ws + (size_t)BATCH * SEQ * DIM);
  unsigned* pkLo  = pkHi + (size_t)NTILES * KSTEPS * 32 * 8;

  // 1) pack Wr into WMMA B fragments (hi/lo bf16)
  pack_wr_kernel<<<(NTILES * KSTEPS * 32 * 8) / 256, 256, 0, stream>>>(Wr, pkHi, pkLo);

  // 2) sequential ODE scan, one block per batch chain
  scan_kernel<<<BATCH, 256, 0, stream>>>(ids, emb, x0, v0, lnG, lnB, U, V, Wc,
                                         roG, roB, outLN, xv);

  // 3) split-bf16 WMMA GEMM for logits
  dim3 grid(32, 250);
  size_t lds = 2u * 16384u * sizeof(unsigned);   // 128 KB A fragments
  gemm_kernel<<<grid, 256, lds, stream>>>(outLN, pkHi, pkLo, br, out);
}
